// Qwen3VLMoeTextSparseMoeBlockTransposed_9775345566133
// MI455X (gfx1250) — matrix operand order repaired and checked by reading, not compiled, not run on hardware
//
#include <hip/hip_runtime.h>

typedef __attribute__((ext_vector_type(8)))  float   v8f;
typedef __attribute__((ext_vector_type(16))) __bf16  v16bf;
typedef __attribute__((ext_vector_type(8)))  __bf16  v8bf;
typedef __attribute__((ext_vector_type(4)))  unsigned int v4u;
typedef __attribute__((ext_vector_type(8)))  int     v8i_;
typedef __attribute__((ext_vector_type(4)))  int     v4i_;

#define T_TOK 2048
#define HDIM  2048
#define NEXP  64
#define IDIM  768
#define TOPK  8
#define CAP   1024
#define LDA   40   // padded LDS pitch (bf16 elements): 32 data + 4-DWORD pad

// workspace layout (bytes)
#define WS_COUNTS   0
#define WS_SLOT_TOK 1024
#define WS_SLOT_W   (1024 + NEXP*CAP*4)
#define WS_H        (1024 + 2*NEXP*CAP*4)

// ---------------------------------------------------------------------------
// WMMA fragment loaders (wave32, 16x16x32 bf16), per CDNA5 ISA VGPR layouts.
// A tile in LDS: row-major [M][LDA]. Lane L (0..15): M=L, K={0..7,16..23};
// lane L (16..31): M=L-16, K={8..15,24..31}.
// B tile in LDS: stored transposed [n][LDA]: lanes 0..15 -> N=lane, K={0..7,16..23};
// lanes 16..31 -> N=lane-16, K={8..15,24..31}.
// ---------------------------------------------------------------------------
__device__ __forceinline__ v16bf frag_a(const __bf16* base, int row0) {
  int lane = threadIdx.x & 31;
  int r  = row0 + (lane & 15);
  int kb = (lane & 16) ? 8 : 0;
  const __bf16* p = base + r * LDA + kb;
  v8bf lo = *(const v8bf*)(p);
  v8bf hi = *(const v8bf*)(p + 16);
  return __builtin_shufflevector(lo, hi, 0,1,2,3,4,5,6,7,8,9,10,11,12,13,14,15);
}

__device__ __forceinline__ v16bf frag_b(const __bf16* base, int col0) {
  int lane = threadIdx.x & 31;
  int n  = col0 + (lane & 15);
  int kb = (lane & 16) ? 8 : 0;
  const __bf16* p = base + n * LDA + kb;
  v8bf lo = *(const v8bf*)(p);
  v8bf hi = *(const v8bf*)(p + 16);
  return __builtin_shufflevector(lo, hi, 0,1,2,3,4,5,6,7,8,9,10,11,12,13,14,15);
}

#define WMMA_BF16(A, B, C) \
  __builtin_amdgcn_wmma_f32_16x16x32_bf16(false, (A), false, (B), (short)0, (C), false, false)

// ---------------------------------------------------------------------------
// TDM: issue tensor_load_to_lds for a [tile_rows x 32] bf16 2D tile with
// row stride `row_elems`, LDS-padded to pitch LDA (pad 4 DWORDs per 16 DWORDs).
// Dynamic words are forced scalar with readfirstlane (D# is SGPR-sourced).
// This toolchain's builtin takes 6 args: (v4u, v8i, v4i, v4i, v8i, i32 cpol).
// ---------------------------------------------------------------------------
__device__ __forceinline__ void tdm_load_tile_bf16(unsigned lds_off,
                                                   unsigned long long gaddr,
                                                   unsigned row_elems,
                                                   unsigned tile_rows) {
  unsigned g_lo = (unsigned)__builtin_amdgcn_readfirstlane((int)(gaddr & 0xffffffffu));
  unsigned g_hi = (unsigned)__builtin_amdgcn_readfirstlane((int)((gaddr >> 32) & 0x1ffffffu));
  unsigned lds  = (unsigned)__builtin_amdgcn_readfirstlane((int)lds_off);

  // group0: count=1 | lds_addr | global_addr | type=2 (bits 127:126)
  v4u g0 = { 1u, lds, g_lo, g_hi | 0x80000000u };

  // group1:
  // w0: data_size=1 (2B) <<16 | pad_enable<<20 | pad_interval=3 (16 DW) <<22
  //     | pad_amount=3 (4 DW) <<25
  // tensor_dim0 (bits 79:48) = row_elems ; tensor_dim1 (111:80) = tile_rows
  // tile_dim0 (127:112) = 32 ; tile_dim1 (143:128) = tile_rows
  // tensor_dim0_stride (207:160) = row_elems
  int w0 = (int)((1u << 16) | (1u << 20) | (3u << 22) | (3u << 25));
  int w1 = (int)((row_elems & 0xffffu) << 16);
  int w2 = (int)(((row_elems >> 16) & 0xffffu) | ((tile_rows & 0xffffu) << 16));
  int w3 = (int)(((tile_rows >> 16) & 0xffffu) | (32u << 16));
  int w4 = (int)(tile_rows & 0xffffu);
  int w5 = (int)row_elems;
  v8i_ g1 = { w0, w1, w2, w3, w4, w5, 0, 0 };

  v4i_ gz4 = { 0, 0, 0, 0 };
  v8i_ gz8 = { 0, 0, 0, 0, 0, 0, 0, 0 };
  __builtin_amdgcn_tensor_load_to_lds(g0, g1, gz4, gz4, gz8, 0);
}

// ---------------------------------------------------------------------------
// Kernel 0: zero output (we accumulate with atomics) and expert counts.
// ---------------------------------------------------------------------------
__global__ void moe_zero_kernel(float* __restrict__ out, size_t n,
                                int* __restrict__ counts) {
  size_t i = (size_t)blockIdx.x * blockDim.x + threadIdx.x;
  size_t stride = (size_t)gridDim.x * blockDim.x;
  for (; i < n; i += stride) out[i] = 0.0f;
  if (blockIdx.x == 0 && threadIdx.x < NEXP) counts[threadIdx.x] = 0;
}

// ---------------------------------------------------------------------------
// Kernel 1: router. One block (64 threads) per token.
// softmax over 64 experts, top-8, renormalize, atomic dispatch into slots.
// ---------------------------------------------------------------------------
__global__ __launch_bounds__(64)
void moe_router_kernel(const float* __restrict__ x,
                       const float* __restrict__ rw,
                       int* __restrict__ counts,
                       int* __restrict__ slot_tok,
                       float* __restrict__ slot_w) {
  __shared__ float xs[HDIM];
  __shared__ float probs[NEXP];
  __shared__ float redv[NEXP];
  __shared__ int   redi[NEXP];
  __shared__ float topw_s[TOPK];
  __shared__ int   topi_s[TOPK];

  const int t   = blockIdx.x;
  const int tid = threadIdx.x;  // 0..63 : one expert per thread

  for (int j = 0; j < HDIM / 64; ++j)
    xs[j * 64 + tid] = x[(size_t)t * HDIM + j * 64 + tid];
  __syncthreads();

  const float4* w4 = (const float4*)(rw + (size_t)tid * HDIM);
  float acc = 0.0f;
  for (int h = 0; h < HDIM / 4; ++h) {
    float4 w = w4[h];
    acc = fmaf(xs[h*4+0], w.x, acc);
    acc = fmaf(xs[h*4+1], w.y, acc);
    acc = fmaf(xs[h*4+2], w.z, acc);
    acc = fmaf(xs[h*4+3], w.w, acc);
  }

  redv[tid] = acc; __syncthreads();
  for (int s = 32; s > 0; s >>= 1) {
    if (tid < s) redv[tid] = fmaxf(redv[tid], redv[tid + s]);
    __syncthreads();
  }
  float mx = redv[0]; __syncthreads();
  float p = __expf(acc - mx);
  redv[tid] = p; __syncthreads();
  for (int s = 32; s > 0; s >>= 1) {
    if (tid < s) redv[tid] += redv[tid + s];
    __syncthreads();
  }
  float inv = 1.0f / redv[0]; __syncthreads();
  probs[tid] = p * inv; __syncthreads();

  for (int k = 0; k < TOPK; ++k) {
    redv[tid] = probs[tid]; redi[tid] = tid; __syncthreads();
    for (int s = 32; s > 0; s >>= 1) {
      if (tid < s && redv[tid + s] > redv[tid]) {
        redv[tid] = redv[tid + s]; redi[tid] = redi[tid + s];
      }
      __syncthreads();
    }
    if (tid == 0) {
      topw_s[k] = redv[0]; topi_s[k] = redi[0];
      probs[redi[0]] = -1.0f;
    }
    __syncthreads();
  }

  if (tid == 0) {
    float s = 0.0f;
    for (int k = 0; k < TOPK; ++k) s += topw_s[k];
    float rinv = 1.0f / s;
    for (int k = 0; k < TOPK; ++k) {
      int e = topi_s[k];
      int pos = atomicAdd(&counts[e], 1);
      if (pos < CAP) {
        slot_tok[e * CAP + pos] = t;
        slot_w[e * CAP + pos]   = topw_s[k] * rinv;
      }
    }
  }
}

// ---------------------------------------------------------------------------
// Kernel 2: gate_up GEMM + SiLU*u, h -> bf16 scratch.
// grid = (IDIM/64, CAP/256, NEXP), block = 512 (16 waves, 8x2 wave tiling).
// Double-buffered LDS + register prefetch: one barrier per K-step, next tile's
// global loads overlap the current tile's WMMA.
// ---------------------------------------------------------------------------
__global__ __launch_bounds__(512)
void moe_gateup_kernel(const float* __restrict__ x,
                       const float* __restrict__ gup,
                       const int*   __restrict__ counts,
                       const int*   __restrict__ slot_tok,
                       __bf16*      __restrict__ hbuf) {
  const int e = blockIdx.z;
  int cnt = counts[e]; if (cnt > CAP) cnt = CAP;
  const int m0 = blockIdx.y * 256;
  if (m0 >= cnt) return;
  int rows = cnt - m0; if (rows > 256) rows = 256;
  const int n0 = blockIdx.x * 64;

  __shared__ __bf16 sA [2][256 * LDA];
  __shared__ __bf16 sBg[2][64 * LDA];
  __shared__ __bf16 sBu[2][64 * LDA];

  const int tid  = threadIdx.x;
  const int lane = tid & 31;
  const int wm   = (tid >> 5) >> 1;   // 0..7 (32-row band)
  const int wn   = (tid >> 5) & 1;    // 0..1 (32-col band)

  // staging coords: A gather (one half-row of 16 per thread)
  const int ar  = tid >> 1;           // 0..255
  const int akh = (tid & 1) << 4;
  const float* asrc = nullptr;
  if (ar < rows)
    asrc = x + (size_t)slot_tok[e * CAP + m0 + ar] * HDIM + akh;
  // B staging: 4 contiguous columns per thread
  const int bk = tid >> 4;            // 0..31
  const int bn = (tid & 15) << 2;     // 0..60
  const float* gbase = gup + (size_t)e * HDIM * (2 * IDIM);

  v8f gacc[2][2] = {{{},{}},{{},{}}};
  v8f uacc[2][2] = {{{},{}},{{},{}}};

  // register prefetch buffers
  float4 aR0 = {}, aR1 = {}, aR2 = {}, aR3 = {};
  float4 bgR = {}, buR = {};

  // prologue: prefetch tile k0 = 0
  {
    if (asrc) {
      const float4* s4 = (const float4*)(asrc);
      aR0 = s4[0]; aR1 = s4[1]; aR2 = s4[2]; aR3 = s4[3];
    }
    const float* gs = gbase + (size_t)bk * (2 * IDIM) + n0 + bn;
    bgR = ((const float4*)gs)[0];
    buR = ((const float4*)(gs + IDIM))[0];
  }

  int cur = 0;
  for (int k0 = 0; k0 < HDIM; k0 += 32) {
    // ---- store prefetched regs into LDS[cur] (fp32 -> bf16) ----
    {
      __bf16* dst = &sA[cur][ar * LDA + akh];
      if (asrc) {
        float av[16] = {aR0.x,aR0.y,aR0.z,aR0.w, aR1.x,aR1.y,aR1.z,aR1.w,
                        aR2.x,aR2.y,aR2.z,aR2.w, aR3.x,aR3.y,aR3.z,aR3.w};
#pragma unroll
        for (int q = 0; q < 16; ++q) dst[q] = (__bf16)av[q];
      } else {
#pragma unroll
        for (int q = 0; q < 16; ++q) dst[q] = (__bf16)0.0f;
      }
      float gv[4] = {bgR.x, bgR.y, bgR.z, bgR.w};
      float uv[4] = {buR.x, buR.y, buR.z, buR.w};
#pragma unroll
      for (int j = 0; j < 4; ++j) {
        sBg[cur][(bn + j) * LDA + bk] = (__bf16)gv[j];
        sBu[cur][(bn + j) * LDA + bk] = (__bf16)uv[j];
      }
    }
    __syncthreads();

    // ---- prefetch next tile into regs (overlaps WMMA below) ----
    const int kn = k0 + 32;
    if (kn < HDIM) {
      if (asrc) {
        const float4* s4 = (const float4*)(asrc + kn);
        aR0 = s4[0]; aR1 = s4[1]; aR2 = s4[2]; aR3 = s4[3];
      }
      const float* gs = gbase + (size_t)(kn + bk) * (2 * IDIM) + n0 + bn;
      bgR = ((const float4*)gs)[0];
      buR = ((const float4*)(gs + IDIM))[0];
    }

    // ---- compute from LDS[cur] ----
    v16bf a0  = frag_a(&sA[cur][0],  wm * 32);
    v16bf a1  = frag_a(&sA[cur][0],  wm * 32 + 16);
    v16bf bg0 = frag_b(&sBg[cur][0], wn * 32);
    v16bf bg1 = frag_b(&sBg[cur][0], wn * 32 + 16);
    v16bf bu0 = frag_b(&sBu[cur][0], wn * 32);
    v16bf bu1 = frag_b(&sBu[cur][0], wn * 32 + 16);

    gacc[0][0] = WMMA_BF16(a0, bg0, gacc[0][0]);
    gacc[0][1] = WMMA_BF16(a0, bg1, gacc[0][1]);
    gacc[1][0] = WMMA_BF16(a1, bg0, gacc[1][0]);
    gacc[1][1] = WMMA_BF16(a1, bg1, gacc[1][1]);
    uacc[0][0] = WMMA_BF16(a0, bu0, uacc[0][0]);
    uacc[0][1] = WMMA_BF16(a0, bu1, uacc[0][1]);
    uacc[1][0] = WMMA_BF16(a1, bu0, uacc[1][0]);
    uacc[1][1] = WMMA_BF16(a1, bu1, uacc[1][1]);

    cur ^= 1;
  }

  // ---- epilogue: h = silu(g) * u -> bf16 scratch ----
#pragma unroll
  for (int ms = 0; ms < 2; ++ms)
#pragma unroll
    for (int ns = 0; ns < 2; ++ns) {
      v8f g = gacc[ms][ns], u = uacc[ms][ns];
#pragma unroll
      for (int r = 0; r < 8; ++r) {
        int m = wm * 32 + ms * 16 + r + ((lane & 16) ? 8 : 0);
        if (m < rows) {
          float gvf = g[r], uvf = u[r];
          float hv = (gvf / (1.0f + __expf(-gvf))) * uvf;
          int n = wn * 32 + ns * 16 + (lane & 15);
          hbuf[(size_t)(e * CAP + m0 + m) * IDIM + n0 + n] = (__bf16)hv;
        }
      }
    }
}

// ---------------------------------------------------------------------------
// Kernel 3: down GEMM, scale by routing weight, atomic-add into output.
// grid = (HDIM/64, CAP/256, NEXP), block = 512.
// A tile (contiguous bf16 rows of h) is staged by the Tensor Data Mover into
// double-buffered LDS with hardware padding: wave 0 waits TENSORcnt for the
// current tile before the barrier, then issues the next tile's DMA after it,
// so the TDM transfer overlaps the WMMA loop.
// ---------------------------------------------------------------------------
__global__ __launch_bounds__(512)
void moe_down_kernel(const __bf16* __restrict__ hbuf,
                     const float*  __restrict__ down,
                     const int*    __restrict__ counts,
                     const int*    __restrict__ slot_tok,
                     const float*  __restrict__ slot_w,
                     float*        __restrict__ out) {
  const int e = blockIdx.z;
  int cnt = counts[e]; if (cnt > CAP) cnt = CAP;
  const int m0 = blockIdx.y * 256;
  if (m0 >= cnt) return;
  int rows = cnt - m0; if (rows > 256) rows = 256;
  const int n0 = blockIdx.x * 64;

  __shared__ __bf16 sA[2][256 * LDA];
  __shared__ __bf16 sB[2][64 * LDA];

  const int tid  = threadIdx.x;
  const int lane = tid & 31;
  const int wm   = (tid >> 5) >> 1;
  const int wn   = (tid >> 5) & 1;

  const int bk = tid >> 4;
  const int bn = (tid & 15) << 2;
  const float* dbase = down + (size_t)e * IDIM * HDIM;

  const unsigned sA_lds0 = (unsigned)(size_t)(&sA[0][0]);
  const unsigned sA_lds1 = (unsigned)(size_t)(&sA[1][0]);
  const unsigned long long abase =
      (unsigned long long)(size_t)(hbuf + (size_t)(e * CAP + m0) * IDIM);

  v8f acc[2][2] = {{{},{}},{{},{}}};

  float4 bR = {};

  // prologue: TDM for tile 0 into buffer 0; prefetch B regs for tile 0
  if (tid < 32)
    tdm_load_tile_bf16(sA_lds0, abase, /*row_elems=*/IDIM, /*tile_rows=*/256u);
  {
    const float* ds_ = dbase + (size_t)bk * HDIM + n0 + bn;
    bR = ((const float4*)ds_)[0];
  }

  int cur = 0;
  for (int k0 = 0; k0 < IDIM; k0 += 32) {
    // ---- store prefetched B regs -> LDS[cur] (fp32 -> bf16) ----
    {
      float dv[4] = {bR.x, bR.y, bR.z, bR.w};
#pragma unroll
      for (int j = 0; j < 4; ++j)
        sB[cur][(bn + j) * LDA + bk] = (__bf16)dv[j];
    }
    // current A tile's TDM must be complete before the barrier publishes it
    if (tid < 32) __builtin_amdgcn_s_wait_tensorcnt(0);
    __syncthreads();

    const int kn = k0 + 32;
    // ---- issue next A tile DMA into the other buffer (overlaps compute) ----
    if (kn < IDIM && tid < 32) {
      tdm_load_tile_bf16(cur ? sA_lds0 : sA_lds1,
                         abase + (unsigned)kn * 2u,
                         /*row_elems=*/IDIM, /*tile_rows=*/256u);
    }
    // ---- prefetch next B tile into regs ----
    if (kn < IDIM) {
      const float* ds_ = dbase + (size_t)(kn + bk) * HDIM + n0 + bn;
      bR = ((const float4*)ds_)[0];
    }

    // ---- compute from LDS[cur] ----
    v16bf a0 = frag_a(&sA[cur][0], wm * 32);
    v16bf a1 = frag_a(&sA[cur][0], wm * 32 + 16);
    v16bf b0 = frag_b(&sB[cur][0], wn * 32);
    v16bf b1 = frag_b(&sB[cur][0], wn * 32 + 16);

    acc[0][0] = WMMA_BF16(a0, b0, acc[0][0]);
    acc[0][1] = WMMA_BF16(a0, b1, acc[0][1]);
    acc[1][0] = WMMA_BF16(a1, b0, acc[1][0]);
    acc[1][1] = WMMA_BF16(a1, b1, acc[1][1]);

    cur ^= 1;
  }

  // ---- epilogue: weighted atomic accumulation into y ----
#pragma unroll
  for (int ms = 0; ms < 2; ++ms)
#pragma unroll
    for (int ns = 0; ns < 2; ++ns) {
#pragma unroll
      for (int r = 0; r < 8; ++r) {
        int m = wm * 32 + ms * 16 + r + ((lane & 16) ? 8 : 0);
        if (m < rows) {
          int slot = e * CAP + m0 + m;
          int t = slot_tok[slot];
          float w = slot_w[slot];
          int n = wn * 32 + ns * 16 + (lane & 15);
          atomicAdd(&out[(size_t)t * HDIM + n0 + n], acc[ms][ns][r] * w);
        }
      }
    }
}

// ---------------------------------------------------------------------------
extern "C" void kernel_launch(void* const* d_in, const int* in_sizes, int n_in,
                              void* d_out, int out_size, void* d_ws, size_t ws_size,
                              hipStream_t stream) {
  const float* x    = (const float*)d_in[0];   // [2,1024,2048]
  const float* rw   = (const float*)d_in[1];   // [64,2048]
  const float* gup  = (const float*)d_in[2];   // [64,2048,1536]
  const float* down = (const float*)d_in[3];   // [64,768,2048]
  float* out = (float*)d_out;                  // [2,1024,2048] fp32

  char* ws = (char*)d_ws;
  int*    counts   = (int*)(ws + WS_COUNTS);
  int*    slot_tok = (int*)(ws + WS_SLOT_TOK);
  float*  slot_w   = (float*)(ws + WS_SLOT_W);
  __bf16* hbuf     = (__bf16*)(ws + WS_H);

  moe_zero_kernel<<<2048, 256, 0, stream>>>(out, (size_t)T_TOK * HDIM, counts);
  moe_router_kernel<<<T_TOK, 64, 0, stream>>>(x, rw, counts, slot_tok, slot_w);
  moe_gateup_kernel<<<dim3(IDIM / 64, CAP / 256, NEXP), 512, 0, stream>>>(
      x, gup, counts, slot_tok, hbuf);
  moe_down_kernel<<<dim3(HDIM / 64, CAP / 256, NEXP), 512, 0, stream>>>(
      hbuf, down, counts, slot_tok, slot_w, out);
}
